// ProgressiveLayeredExtractionModel_80367428043115
// MI455X (gfx1250) — compile-verified, matrix operand order checked by reading, not run on hardware
//
#include <hip/hip_runtime.h>
#include <math.h>

// ---------------------------------------------------------------------------
// Types for CDNA5 WMMA (wave32): v_wmma_f32_16x16x32_bf16
// ---------------------------------------------------------------------------
typedef __bf16 bf16_t;
typedef __attribute__((ext_vector_type(16))) __bf16 v16bf;
typedef __attribute__((ext_vector_type(8)))  __bf16 v8bf;
typedef __attribute__((ext_vector_type(8)))  float  v8f;
typedef __attribute__((ext_vector_type(4)))  int    i4v;

// Toolchain feature probes ---------------------------------------------------
#if defined(__has_builtin)
#  if __has_builtin(__builtin_amdgcn_global_load_async_to_lds_b128)
#    define HAVE_ASYNC_COPY 1
#  endif
#  if __has_builtin(__builtin_amdgcn_s_wait_asynccnt)
#    define HAVE_WAIT_ASYNC 1
#  endif
#endif

// Async copy builtin takes int4 pointers: (as1 src, as3 dst, imm off, imm cpol)
typedef __attribute__((address_space(1))) i4v* g_i4p;
typedef __attribute__((address_space(3))) i4v* l_i4p;

static __device__ __forceinline__ g_i4p to_as1_i4(const void* p) {
  return (g_i4p)(unsigned long long)p;
}
static __device__ __forceinline__ l_i4p to_as3_i4(const void* p) {
  // generic LDS pointer's low 32 bits are the LDS byte offset
  return (l_i4p)(unsigned int)(unsigned long long)p;
}

// Fragment loader for the 16-bit A/B VGPR striping (ISA 7.12.2):
//   lane L: row = L&15, hi = L>>4
//   elements 0..7  <-> K = k0 + hi*8 + e        (contiguous 16 bytes)
//   elements 8..15 <-> K = k0 + 16 + hi*8 + e   (contiguous 16 bytes, +32B)
// Works for global (global_load_b128) and LDS (ds_load_b128) pointers.
static __device__ __forceinline__ v16bf frag_load(const bf16_t* p) {
  union { v16bf v; v8bf h[2]; } u;
  u.h[0] = *reinterpret_cast<const v8bf*>(p);
  u.h[1] = *reinterpret_cast<const v8bf*>(p + 16);
  return u.v;
}

// ---------------------------------------------------------------------------
// C = relu(A[MxK] @ W[KxN] + bias[N]); A bf16 row-major, W given transposed
// (WT bf16, [N x K] row-major), C bf16 row-major.
// Block = 256 threads (8 waves). Block tile 128(M) x 64(N). B tile staged in
// LDS (double-buffered, K-chunk 64, padded row stride 72 elems) via async
// global->LDS copies when available; each wave owns a 16x64 strip.
// Requires: M % 128 == 0, N % 64 == 0, K % 64 == 0 (true for all layers here).
// ---------------------------------------------------------------------------
#define BK        64   // K elements per stage
#define LDS_STRIDE 72  // padded row stride in elements (144 B -> 36 banks)

__global__ __launch_bounds__(256) void wmma_gemm_bias_relu(
    const bf16_t* __restrict__ A,
    const bf16_t* __restrict__ WT,
    const float*  __restrict__ bias,
    bf16_t* __restrict__ C,
    int N, int K)
{
  __shared__ bf16_t Bt[2][64 * LDS_STRIDE];   // 2 x 9216 B = 18432 B

  const int tid  = threadIdx.x;
  const int lane = tid & 31;
  const int wave = tid >> 5;                  // 0..7
  const int m0   = blockIdx.y * 128 + wave * 16;
  const int n0   = blockIdx.x * 64;
  const int r    = lane & 15;
  const int hi   = lane >> 4;

  // Staging assignment: 512 x 16B segments per stage, 2 per thread.
  const int row0 = tid >> 3;                  // 0..31 (and +32)
  const int kk0  = (tid & 7) * 8;             // 0..56

  const bf16_t* g0base = WT + (size_t)(n0 + row0) * K + kk0;
  const bf16_t* g1base = WT + (size_t)(n0 + row0 + 32) * K + kk0;
  bf16_t* l0[2] = { &Bt[0][row0 * LDS_STRIDE + kk0],
                    &Bt[1][row0 * LDS_STRIDE + kk0] };
  bf16_t* l1[2] = { &Bt[0][(row0 + 32) * LDS_STRIDE + kk0],
                    &Bt[1][(row0 + 32) * LDS_STRIDE + kk0] };

  const bf16_t* aP = A + (size_t)(m0 + r) * K + hi * 8;

  v8f acc[4] = {v8f{}, v8f{}, v8f{}, v8f{}};
  const int nkb = K / BK;

  auto stage = [&](int buf, int k0) {
#if HAVE_ASYNC_COPY
    __builtin_amdgcn_global_load_async_to_lds_b128(to_as1_i4(g0base + k0),
                                                   to_as3_i4(l0[buf]), 0, 0);
    __builtin_amdgcn_global_load_async_to_lds_b128(to_as1_i4(g1base + k0),
                                                   to_as3_i4(l1[buf]), 0, 0);
#else
    v8bf d0 = *reinterpret_cast<const v8bf*>(g0base + k0);
    v8bf d1 = *reinterpret_cast<const v8bf*>(g1base + k0);
    *reinterpret_cast<v8bf*>(l0[buf]) = d0;
    *reinterpret_cast<v8bf*>(l1[buf]) = d1;
#endif
  };

  auto wait_async = [&]() {
#if HAVE_ASYNC_COPY
#  if HAVE_WAIT_ASYNC
    __builtin_amdgcn_s_wait_asynccnt(0);
#  else
    asm volatile("s_wait_asynccnt 0x0" ::: "memory");
#  endif
#endif
  };

  // Prologue: fill buffer 0.
  stage(0, 0);
  wait_async();
  __syncthreads();

  for (int kb = 0; kb < nkb; ++kb) {
    const int cur = kb & 1;
    if (kb + 1 < nkb) stage((kb + 1) & 1, (kb + 1) * BK);
    if (kb + 2 < nkb) __builtin_prefetch(aP + (kb + 2) * BK, 0, 3);

    const int kg = kb * BK;
#pragma unroll
    for (int kc = 0; kc < BK; kc += 32) {
      v16bf a = frag_load(aP + kg + kc);
#pragma unroll
      for (int j = 0; j < 4; ++j) {
        const bf16_t* lp = &Bt[cur][(j * 16 + r) * LDS_STRIDE + kc + hi * 8];
        v16bf b = frag_load(lp);
        acc[j] = __builtin_amdgcn_wmma_f32_16x16x32_bf16(
            /*neg_a=*/false, a, /*neg_b=*/false, b,
            /*c_mod=*/(short)0, acc[j], /*reuse_a=*/false, /*reuse_b=*/false);
      }
    }

    if (kb + 1 < nkb) {
      wait_async();      // next buffer's copies (and fallback ds stores) done
      __syncthreads();   // all waves done reading buffer being overwritten next
    }
  }

  // C/D layout: VGPR i, lane L -> M = m0 + (L>>4)*8 + i, N = n0 + (L&15)
#pragma unroll
  for (int j = 0; j < 4; ++j) {
    const int n  = n0 + j * 16 + r;
    const float bv = bias[n];
#pragma unroll
    for (int i = 0; i < 8; ++i) {
      const int m = m0 + hi * 8 + i;
      float v = acc[j][i] + bv;
      v = v > 0.0f ? v : 0.0f;
      C[(size_t)m * N + n] = (bf16_t)v;
    }
  }
}

// ---------------------------------------------------------------------------
// Embedding gather: ex[b, f*64+e] = bf16(table[(x[b,f] + f*V)*64 + e])
// ---------------------------------------------------------------------------
__global__ void embed_gather(const int* __restrict__ x,
                             const float* __restrict__ table,
                             bf16_t* __restrict__ ex, long total)
{
  long i = (long)blockIdx.x * blockDim.x + threadIdx.x;
  if (i >= total) return;
  int e  = (int)(i & 63);
  long bf = i >> 6;
  int f  = (int)(bf & 31);
  long b = bf >> 5;
  long idx = (long)x[b * 32 + f] + (long)f * 10000;
  ex[i] = (bf16_t)table[idx * 64 + e];
}

// ---------------------------------------------------------------------------
// Batched f32 [nmat,K,N] -> bf16 transposed [nmat,N,K]
// ---------------------------------------------------------------------------
__global__ void transpose_f32_to_bf16(const float* __restrict__ W,
                                      bf16_t* __restrict__ WT,
                                      int K, int N, long total)
{
  long i = (long)blockIdx.x * blockDim.x + threadIdx.x;
  if (i >= total) return;
  long kn  = (long)K * N;
  long mat = i / kn;
  long rem = i - mat * kn;
  int k = (int)(rem / N);
  int n = (int)(rem - (long)k * N);
  WT[mat * kn + (long)n * K + k] = (bf16_t)W[i];
}

// ---------------------------------------------------------------------------
// Tiny gate head: logits = gh[B,256] @ Wf[256,NE]; softmax over NE (<=6)
// ---------------------------------------------------------------------------
__global__ void gate_softmax(const bf16_t* __restrict__ gh,
                             const float* __restrict__ Wf,
                             float* __restrict__ gw, int NE, int Bn)
{
  int b = blockIdx.x * blockDim.x + threadIdx.x;
  if (b >= Bn) return;
  float logit[6];
  for (int n = 0; n < NE; ++n) logit[n] = 0.0f;
  const bf16_t* g = gh + (size_t)b * 256;
  for (int k = 0; k < 256; ++k) {
    float gv = (float)g[k];
    for (int n = 0; n < NE; ++n) logit[n] += gv * Wf[k * NE + n];
  }
  float mx = logit[0];
  for (int n = 1; n < NE; ++n) mx = fmaxf(mx, logit[n]);
  float s = 0.0f;
  for (int n = 0; n < NE; ++n) { logit[n] = __expf(logit[n] - mx); s += logit[n]; }
  float inv = 1.0f / s;
  for (int n = 0; n < NE; ++n) gw[(size_t)b * NE + n] = logit[n] * inv;
}

// ---------------------------------------------------------------------------
// Task combine: out[b,h] = sum_n gw[b,n] * expert_n[b,h]
//   n=0,1 -> s2_t[e=0,1]; n=2,3 -> r2[e=0,1]      (bf16 in, bf16 out)
// ---------------------------------------------------------------------------
__global__ void combine_task(const float* __restrict__ gw,
                             const bf16_t* __restrict__ s2_t,
                             const bf16_t* __restrict__ r2,
                             bf16_t* __restrict__ out, long BH)
{
  long i = (long)blockIdx.x * blockDim.x + threadIdx.x;
  if (i >= BH) return;
  long b = i >> 9;  // H2 = 512
  const float* w = gw + b * 4;
  float v = w[0] * (float)s2_t[i]
          + w[1] * (float)s2_t[BH + i]
          + w[2] * (float)r2[i]
          + w[3] * (float)r2[BH + i];
  out[i] = (bf16_t)v;
}

// Shared combine over 6 experts: order [t0e0,t0e1,t1e0,t1e1,r0,r1]
__global__ void combine_shared(const float* __restrict__ hgw,
                               const bf16_t* __restrict__ s2,
                               const bf16_t* __restrict__ r2,
                               bf16_t* __restrict__ out, long BH)
{
  long i = (long)blockIdx.x * blockDim.x + threadIdx.x;
  if (i >= BH) return;
  long b = i >> 9;
  const float* w = hgw + b * 6;
  float v = w[0] * (float)s2[i]
          + w[1] * (float)s2[BH + i]
          + w[2] * (float)s2[2 * BH + i]
          + w[3] * (float)s2[3 * BH + i]
          + w[4] * (float)r2[i]
          + w[5] * (float)r2[BH + i];
  out[i] = (bf16_t)v;
}

// ---------------------------------------------------------------------------
// Tower head: out[b*T+t] = sigmoid( th[t][b,:] . tow_Wf[t] )
// ---------------------------------------------------------------------------
__global__ void tower_out_kernel(const bf16_t* __restrict__ th,
                                 const float* __restrict__ Wf,
                                 float* __restrict__ out, int Bn)
{
  int i = blockIdx.x * blockDim.x + threadIdx.x;
  if (i >= Bn * 2) return;
  int t = i & 1;
  int b = i >> 1;
  const bf16_t* h = th + (size_t)t * Bn * 256 + (size_t)b * 256;
  const float*  w = Wf + t * 256;
  float s = 0.0f;
  for (int k = 0; k < 256; ++k) s += (float)h[k] * w[k];
  out[(size_t)b * 2 + t] = 1.0f / (1.0f + __expf(-s));
}

// ---------------------------------------------------------------------------
// Host orchestration
// ---------------------------------------------------------------------------
extern "C" void kernel_launch(void* const* d_in, const int* in_sizes, int n_in,
                              void* d_out, int out_size, void* d_ws, size_t ws_size,
                              hipStream_t stream) {
  constexpr int B = 16384;
  constexpr int D0 = 2048, H1 = 1024, H2 = 512, G = 256, TW = 256;
  constexpr int T = 2, ES = 2, ESH = 2;
  const int dins[2] = {D0, H2};

  const int*   x     = (const int*)d_in[0];
  const float* embed = (const float*)d_in[1];
  const float* spec_W1[2] = {(const float*)d_in[2],  (const float*)d_in[3]};
  const float* spec_b1[2] = {(const float*)d_in[4],  (const float*)d_in[5]};
  const float* spec_W2[2] = {(const float*)d_in[6],  (const float*)d_in[7]};
  const float* spec_b2[2] = {(const float*)d_in[8],  (const float*)d_in[9]};
  const float* shr_W1[2]  = {(const float*)d_in[10], (const float*)d_in[11]};
  const float* shr_b1[2]  = {(const float*)d_in[12], (const float*)d_in[13]};
  const float* shr_W2[2]  = {(const float*)d_in[14], (const float*)d_in[15]};
  const float* shr_b2[2]  = {(const float*)d_in[16], (const float*)d_in[17]};
  const float* sg_W1[2]   = {(const float*)d_in[18], (const float*)d_in[19]};
  const float* sg_b1[2]   = {(const float*)d_in[20], (const float*)d_in[21]};
  const float* sg_Wf[2]   = {(const float*)d_in[22], (const float*)d_in[23]};
  const float* hg_W1[2]   = {(const float*)d_in[24], (const float*)d_in[25]};
  const float* hg_b1[2]   = {(const float*)d_in[26], (const float*)d_in[27]};
  const float* hg_Wf[2]   = {(const float*)d_in[28], (const float*)d_in[29]};
  const float* tow_W1 = (const float*)d_in[30];
  const float* tow_b1 = (const float*)d_in[31];
  const float* tow_Wf = (const float*)d_in[32];
  float* out = (float*)d_out;

  // ---- workspace carve (256B aligned) ----
  char* wp = (char*)d_ws;
  auto carve = [&](size_t bytes) -> void* {
    void* q = (void*)wp;
    wp += (bytes + 255) & ~(size_t)255;
    return q;
  };
  const long BH = (long)B * H2;

  bf16_t* ex     = (bf16_t*)carve((size_t)B * D0 * 2);
  bf16_t* tnext0 = (bf16_t*)carve((size_t)BH * 2);
  bf16_t* tnext1 = (bf16_t*)carve((size_t)BH * 2);
  bf16_t* shnext = (bf16_t*)carve((size_t)BH * 2);
  bf16_t* s1     = (bf16_t*)carve((size_t)B * H1 * 2);          // reused per expert
  bf16_t* s2     = (bf16_t*)carve((size_t)T * ES * BH * 2);     // [t*ES+e][B,H2]
  bf16_t* r2     = (bf16_t*)carve((size_t)ESH * BH * 2);
  bf16_t* gh     = (bf16_t*)carve((size_t)T * B * G * 2);
  bf16_t* hgh    = (bf16_t*)carve((size_t)B * G * 2);
  bf16_t* th     = (bf16_t*)carve((size_t)T * B * TW * 2);
  float*  gw     = (float*)carve((size_t)T * B * 4 * 4);
  float*  hgw    = (float*)carve((size_t)B * 6 * 4);

  bf16_t* specW1T[2], *specW2T[2], *shrW1T[2], *shrW2T[2], *sgW1T[2], *hgW1T[2];
  for (int l = 0; l < 2; ++l) {
    specW1T[l] = (bf16_t*)carve((size_t)T * ES * dins[l] * H1 * 2);
    specW2T[l] = (bf16_t*)carve((size_t)T * ES * H1 * H2 * 2);
    shrW1T[l]  = (bf16_t*)carve((size_t)ESH * dins[l] * H1 * 2);
    shrW2T[l]  = (bf16_t*)carve((size_t)ESH * H1 * H2 * 2);
    sgW1T[l]   = (bf16_t*)carve((size_t)T * dins[l] * G * 2);
    hgW1T[l]   = (bf16_t*)carve((size_t)dins[l] * G * 2);
  }
  bf16_t* towW1T = (bf16_t*)carve((size_t)T * H2 * TW * 2);

  // ---- launch helpers ----
  auto tr = [&](const float* W, bf16_t* WT, int K, int N, int nmat) {
    long total = (long)nmat * K * N;
    int blocks = (int)((total + 255) / 256);
    transpose_f32_to_bf16<<<blocks, 256, 0, stream>>>(W, WT, K, N, total);
  };
  auto gemm = [&](const bf16_t* A, const bf16_t* WT, const float* bias,
                  bf16_t* C, int N, int K) {
    dim3 grid(N / 64, B / 128);
    wmma_gemm_bias_relu<<<grid, dim3(256), 0, stream>>>(A, WT, bias, C, N, K);
  };

  // ---- 1) embedding gather ----
  {
    long total = (long)B * D0;
    embed_gather<<<(int)((total + 255) / 256), 256, 0, stream>>>(x, embed, ex, total);
  }

  // ---- 2) weight conversions (f32 -> bf16, transposed to [N,K]) ----
  for (int l = 0; l < 2; ++l) {
    tr(spec_W1[l], specW1T[l], dins[l], H1, T * ES);
    tr(spec_W2[l], specW2T[l], H1, H2, T * ES);
    tr(shr_W1[l],  shrW1T[l],  dins[l], H1, ESH);
    tr(shr_W2[l],  shrW2T[l],  H1, H2, ESH);
    tr(sg_W1[l],   sgW1T[l],   dins[l], G, T);
    tr(hg_W1[l],   hgW1T[l],   dins[l], G, 1);
  }
  tr(tow_W1, towW1T, H2, TW, T);

  // ---- 3) PLE levels ----
  const bf16_t* tin[2] = {ex, ex};
  const bf16_t* sh = ex;
  const int blocksBH = (int)((BH + 255) / 256);
  const int blocksB  = (B + 255) / 256;

  for (int l = 0; l < 2; ++l) {
    const int din = dins[l];
    // specific experts (2-layer MLP, relu)
    for (int t = 0; t < T; ++t)
      for (int e = 0; e < ES; ++e) {
        const int te = t * ES + e;
        gemm(tin[t], specW1T[l] + (size_t)te * H1 * din,
             spec_b1[l] + (size_t)te * H1, s1, H1, din);
        gemm(s1, specW2T[l] + (size_t)te * H2 * H1,
             spec_b2[l] + (size_t)te * H2, s2 + (size_t)te * BH, H2, H1);
      }
    // shared experts
    for (int e = 0; e < ESH; ++e) {
      gemm(sh, shrW1T[l] + (size_t)e * H1 * din,
           shr_b1[l] + (size_t)e * H1, s1, H1, din);
      gemm(s1, shrW2T[l] + (size_t)e * H2 * H1,
           shr_b2[l] + (size_t)e * H2, r2 + (size_t)e * BH, H2, H1);
    }
    // per-task gates
    for (int t = 0; t < T; ++t) {
      gemm(tin[t], sgW1T[l] + (size_t)t * G * din,
           sg_b1[l] + (size_t)t * G, gh + (size_t)t * B * G, G, din);
      gate_softmax<<<blocksB, 256, 0, stream>>>(
          gh + (size_t)t * B * G, sg_Wf[l] + (size_t)t * G * (ES + ESH),
          gw + (size_t)t * B * 4, ES + ESH, B);
    }
    // task combine -> next-level task inputs
    bf16_t* tdst[2] = {tnext0, tnext1};
    for (int t = 0; t < T; ++t)
      combine_task<<<blocksBH, 256, 0, stream>>>(
          gw + (size_t)t * B * 4, s2 + (size_t)t * ES * BH, r2, tdst[t], BH);
    // shared gate + combine (only needed when another level consumes it)
    if (l == 0) {
      gemm(sh, hgW1T[l], hg_b1[l], hgh, G, din);
      gate_softmax<<<blocksB, 256, 0, stream>>>(hgh, hg_Wf[l], hgw, T * ES + ESH, B);
      combine_shared<<<blocksBH, 256, 0, stream>>>(hgw, s2, r2, shnext, BH);
      sh = shnext;
    }
    tin[0] = tnext0;
    tin[1] = tnext1;
  }

  // ---- 4) towers + sigmoid ----
  for (int t = 0; t < T; ++t)
    gemm(tin[t], towW1T + (size_t)t * TW * H2,
         tow_b1 + (size_t)t * TW, th + (size_t)t * B * TW, TW, H2);
  tower_out_kernel<<<(B * 2 + 255) / 256, 256, 0, stream>>>(th, tow_Wf, out, B);
}
